// TinyKNN_7086696038575
// MI455X (gfx1250) — matrix operand order repaired and checked by reading, not compiled, not run on hardware
//
#include <hip/hip_runtime.h>
#include <math.h>

// ---------------------------------------------------------------------------
// CDNA5 (gfx1250) wave32 WMMA implementation of the point-cloud network.
// bf16 v_wmma_f32_16x16x32_bf16 for all heavy GEMMs; BN stats computed
// analytically via first/second moments (14x14 and 64x64 Grams).
// All WMMA fragments are staged in *fragment order* so each lane loads its
// 32B fragment with wide b128 loads (no per-element ds_load_u16 assembly).
// ---------------------------------------------------------------------------

typedef __attribute__((ext_vector_type(16))) __bf16 v16bf;
typedef __attribute__((ext_vector_type(8)))  float  v8f;

#define DEVFN static __device__ __forceinline__

DEVFN unsigned short f2bfu(float f) {
  unsigned u = __builtin_bit_cast(unsigned, f);
  return (unsigned short)((u + 0x7fffu + ((u >> 16) & 1u)) >> 16);  // RNE
}
DEVFN float bfu2f(unsigned short s) {
  unsigned u = ((unsigned)s) << 16;
  return __builtin_bit_cast(float, u);
}
DEVFN __bf16 usb(unsigned short s) { return __builtin_bit_cast(__bf16, s); }
DEVFN float gelu_f(float x) { return 0.5f * x * (1.0f + erff(x * 0.70710678118654752f)); }
DEVFN v8f vzero() { v8f z = {0.f,0.f,0.f,0.f,0.f,0.f,0.f,0.f}; return z; }

// 16-bit A-matrix 16x32 lane layout (ISA 7.12.2): element j of lane L maps to
// K = akidx(j, L>>4), M = L&15.
DEVFN int akidx(int j, int hi) {
  int r = j >> 1;
  int base = (r < 4) ? (2 * r) : (16 + 2 * (r - 4));
  return base + 8 * hi + (j & 1);
}
// Inverse: K index kk (0..31) -> (hi, j) of the lane/element holding it:
DEVFN int ahi(int kk) { return (kk & 8) >> 3; }
DEVFN int ajj(int kk) { return (kk & 7) | ((kk >> 1) & 8); }
// 16-bit B-matrix 32x16 as used here (contiguous k, A matches): lane L element j -> K = (L>>4)*16+j, N = L&15.
// f32 C/D 16x16: lane L element r -> M = r + 8*(L>>4), N = L&15.

DEVFN v8f wmma_bf16(v16bf a, v16bf b, v8f c) {
  return __builtin_amdgcn_wmma_f32_16x16x32_bf16(false, a, false, b, (short)0, c, false, false);
}
DEVFN v16bf ld_frag_lds(const unsigned short* p) { return *(const v16bf*)p; }
DEVFN v16bf ld_frag_g(const unsigned short* p) { return *(const v16bf*)p; }

// ---------------------------------------------------------------------------
// K1: 16-NN per cloud (xyz only, cloud resident in LDS)
// ---------------------------------------------------------------------------
__global__ __launch_bounds__(256) void k_knn(const float* __restrict__ x, int* __restrict__ gidx) {
  __shared__ float sx[1024][3];
  __shared__ float ssq[1024];
  int b = blockIdx.x, tid = threadIdx.x;
  const float* xb = x + (size_t)b * 1024 * 7;
  for (int i = tid; i < 1024; i += 256) {
    float a0 = xb[i*7+0], a1 = xb[i*7+1], a2 = xb[i*7+2];
    sx[i][0] = a0; sx[i][1] = a1; sx[i][2] = a2;
    ssq[i] = a0*a0 + a1*a1 + a2*a2;
  }
  __syncthreads();
  for (int q = tid; q < 1024; q += 256) {
    float qx = sx[q][0], qy = sx[q][1], qz = sx[q][2], qs = ssq[q];
    float bd[16]; int bi[16];
    #pragma unroll
    for (int k = 0; k < 16; k++) { bd[k] = 3.4e38f; bi[k] = 0; }
    for (int j = 0; j < 1024; j++) {
      float d = qs + ssq[j] - 2.0f * (qx*sx[j][0] + qy*sx[j][1] + qz*sx[j][2]);
      if (d < bd[15]) {
        int k = 15;
        while (k > 0 && bd[k-1] > d) { bd[k] = bd[k-1]; bi[k] = bi[k-1]; k--; }
        bd[k] = d; bi[k] = j;
      }
    }
    int* op = gidx + ((size_t)b * 1024 + q) * 16;
    #pragma unroll
    for (int k = 0; k < 16; k++) op[k] = bi[k];
  }
}

// ---------------------------------------------------------------------------
// K2: edge-feature moments (Sum e [14], Gram e e^T [14x14]) -> per-WG partials
// ---------------------------------------------------------------------------
__global__ __launch_bounds__(256) void k_stats1(const float* __restrict__ x, const int* __restrict__ gidx,
                                                float* __restrict__ part1) {
  __shared__ float sx[1024][7];
  __shared__ float ech[128][14];
  int b = blockIdx.x, tid = threadIdx.x;
  const float* xb = x + (size_t)b * 1024 * 7;
  for (int i = tid; i < 1024 * 7; i += 256) sx[i / 7][i % 7] = xb[i];
  __syncthreads();
  int i0 = -1, j0 = -1;
  if (tid < 196) { i0 = tid / 14; j0 = tid % 14; }
  else if (tid < 210) { i0 = tid - 196; }
  float acc = 0.f;
  const int* idxb = gidx + (size_t)b * 1024 * 16;
  for (int base = 0; base < 16384; base += 128) {
    int r = tid >> 1, half = tid & 1;
    int row = base + r, p = row >> 4;
    int jn = idxb[row];
    #pragma unroll
    for (int cc = 0; cc < 7; cc++) {
      float v = (half == 0) ? sx[p][cc] : (sx[jn][cc] - sx[p][cc]);
      ech[r][half * 7 + cc] = v;
    }
    __syncthreads();
    if (tid < 196) { for (int r2 = 0; r2 < 128; r2++) acc += ech[r2][i0] * ech[r2][j0]; }
    else if (tid < 210) { for (int r2 = 0; r2 < 128; r2++) acc += ech[r2][i0]; }
    __syncthreads();
  }
  if (tid < 210) part1[b * 210 + tid] = acc;
}

// ---------------------------------------------------------------------------
// K3: finalize BN1
// ---------------------------------------------------------------------------
__global__ __launch_bounds__(256) void k_fin1(const float* __restrict__ part1, const float* __restrict__ w1,
                                              const float* __restrict__ g1, const float* __restrict__ b1,
                                              float* __restrict__ bnp1) {
  __shared__ float st[210];
  int tid = threadIdx.x;
  if (tid < 210) { float s = 0.f; for (int w = 0; w < 64; w++) s += part1[w * 210 + tid]; st[tid] = s; }
  __syncthreads();
  if (tid < 64) {
    const float invN = 1.0f / 1048576.0f;
    float mean = 0.f;
    for (int i = 0; i < 14; i++) mean += w1[tid * 14 + i] * st[196 + i];
    mean *= invN;
    float ex2 = 0.f;
    for (int i = 0; i < 14; i++) {
      float wi = w1[tid * 14 + i];
      for (int j = 0; j < 14; j++) ex2 += wi * w1[tid * 14 + j] * st[i * 14 + j];
    }
    ex2 *= invN;
    float var = ex2 - mean * mean;
    float sc = g1[tid] * rsqrtf(var + 1e-5f);
    bnp1[tid] = sc;
    bnp1[64 + tid] = b1[tid] - mean * sc;
  }
}

// ---------------------------------------------------------------------------
// K4: stream h1 = gelu(bn1(E W1^T)); accumulate Sum h1 (registers) and Gram
// h1^T h1 (WMMA, fragment-order LDS staging). Deterministic per-wave partials.
// ---------------------------------------------------------------------------
__global__ __launch_bounds__(256) void k_stats2(const float* __restrict__ x, const int* __restrict__ gidx,
                                                const float* __restrict__ ecw1, const float* __restrict__ bnp1,
                                                float* __restrict__ part2) {
  __shared__ unsigned short sxb[1024][8];
  __shared__ unsigned short Bw1[32][64];
  __shared__ float sc1[64], sh1[64];
  __shared__ alignas(32) unsigned short Af2[8][4][32][16];  // h1^T in A-frag order
  __shared__ alignas(32) unsigned short Bf2[8][4][32][16];  // h1 in B-frag order
  int b = blockIdx.x, tid = threadIdx.x;
  int wave = tid >> 5, lane = tid & 31, n = lane & 15, hi = lane >> 4;
  const float* xg = x + (size_t)b * 1024 * 7;
  for (int i = tid; i < 1024 * 7; i += 256) sxb[i / 7][i % 7] = f2bfu(xg[i]);
  for (int i = tid; i < 32 * 64; i += 256) {
    int k = i >> 6, o = i & 63;
    Bw1[k][o] = (k < 14) ? f2bfu(ecw1[o * 14 + k]) : (unsigned short)0;
  }
  if (tid < 64) { sc1[tid] = bnp1[tid]; sh1[tid] = bnp1[64 + tid]; }
  __syncthreads();
  v16bf w1f[4];  // hoisted weight fragments
  #pragma unroll
  for (int t = 0; t < 4; t++)
    #pragma unroll
    for (int j = 0; j < 16; j++) w1f[t][j] = usb(Bw1[hi * 16 + j][t * 16 + n]);
  const int* idxb = gidx + (size_t)b * 1024 * 16;
  v8f g[16];
  #pragma unroll
  for (int t = 0; t < 16; t++) g[t] = vzero();
  float srow[4] = {0.f, 0.f, 0.f, 0.f};
  for (int pp = wave * 128; pp < wave * 128 + 128; pp += 2) {
    #pragma unroll
    for (int sub = 0; sub < 2; sub++) {
      int p = pp + sub;
      int jn = idxb[p * 16 + n];  // this lane's neighbor row (M = n)
      v16bf a;
      #pragma unroll
      for (int j = 0; j < 16; j++) {
        int kk = hi * 16 + j;
        unsigned short v;
        if (kk < 7) v = sxb[p][kk];
        else if (kk < 14) v = f2bfu(bfu2f(sxb[jn][kk - 7]) - bfu2f(sxb[p][kk - 7]));
        else v = 0;
        a[j] = usb(v);
      }
      #pragma unroll
      for (int t = 0; t < 4; t++) {
        v8f c = vzero();
        c = wmma_bf16(a, w1f[t], c);
        int ch = t * 16 + n;
        float sc = sc1[ch], sh = sh1[ch];
        #pragma unroll
        for (int r = 0; r < 8; r++) {
          float val = gelu_f(c[r] * sc + sh);
          srow[t] += val;
          unsigned short vb = f2bfu(val);
          int row = sub * 16 + r + 8 * hi;  // row within 32-row chunk
          Af2[wave][t][(ch & 15) + 16 * (row >> 4)][row & 15] = vb;
          Bf2[wave][t][(ch & 15) + 16 * (row >> 4)][row & 15] = vb;
        }
      }
    }
    v16bf b2f[4];
    #pragma unroll
    for (int tj = 0; tj < 4; tj++) b2f[tj] = ld_frag_lds(&Bf2[wave][tj][lane][0]);
    #pragma unroll
    for (int ti = 0; ti < 4; ti++) {
      v16bf a2 = ld_frag_lds(&Af2[wave][ti][lane][0]);
      #pragma unroll
      for (int tj = 0; tj < 4; tj++) g[ti * 4 + tj] = wmma_bf16(a2, b2f[tj], g[ti * 4 + tj]);
    }
  }
  float* pw = part2 + ((size_t)(b * 8 + wave)) * 4160;
  #pragma unroll
  for (int t = 0; t < 4; t++) {
    float tot = srow[t] + __shfl_xor(srow[t], 16);
    if (hi == 0) pw[t * 16 + n] = tot;
  }
  #pragma unroll
  for (int ti = 0; ti < 4; ti++)
    #pragma unroll
    for (int tj = 0; tj < 4; tj++)
      #pragma unroll
      for (int r = 0; r < 8; r++)
        pw[64 + (ti * 16 + r + 8 * hi) * 64 + tj * 16 + n] = g[ti * 4 + tj][r];
}

// ---------------------------------------------------------------------------
// K5: finalize BN2 from the 64-vec + 64x64 Gram
// ---------------------------------------------------------------------------
__global__ __launch_bounds__(256) void k_fin2(const float* __restrict__ part2, const float* __restrict__ w2,
                                              const float* __restrict__ g2v, const float* __restrict__ b2v,
                                              float* __restrict__ bnp2) {
  __shared__ float st[4160];
  int tid = threadIdx.x;
  for (int e = tid; e < 4160; e += 256) {
    float s = 0.f;
    for (int w = 0; w < 512; w++) s += part2[(size_t)w * 4160 + e];
    st[e] = s;
  }
  __syncthreads();
  if (tid < 128) {
    const float invN = 1.0f / 1048576.0f;
    float mean = 0.f;
    for (int c = 0; c < 64; c++) mean += w2[tid * 64 + c] * st[c];
    mean *= invN;
    float ex2 = 0.f;
    for (int ci = 0; ci < 64; ci++) {
      float wi = w2[tid * 64 + ci];
      for (int cj = 0; cj < 64; cj++) ex2 += wi * w2[tid * 64 + cj] * st[64 + ci * 64 + cj];
    }
    ex2 *= invN;
    float var = ex2 - mean * mean;
    float sc = g2v[tid] * rsqrtf(var + 1e-5f);
    bnp2[tid] = sc;
    bnp2[128 + tid] = b2v[tid] - mean * sc;
  }
}

// ---------------------------------------------------------------------------
// K6: main edge-conv pass. All 20 weight fragments live in VGPRs; h1 is
// staged in fragment order so layer-2 A loads are wide b128 LDS reads.
// ---------------------------------------------------------------------------
__global__ __launch_bounds__(256) void k_edge(const float* __restrict__ x, const int* __restrict__ gidx,
                                              const float* __restrict__ ecw1, const float* __restrict__ ecw2,
                                              const float* __restrict__ bnp1, const float* __restrict__ bnp2,
                                              unsigned short* __restrict__ localf) {
  __shared__ unsigned short sxb[1024][8];
  __shared__ unsigned short Bw1[32][64];
  __shared__ unsigned short Bw2[64][128];
  __shared__ float sc1[64], sh1[64], sc2[128], sh2[128];
  __shared__ alignas(32) unsigned short h1f[8][2][32][16];  // h1 in A-frag order
  int b = blockIdx.x, tid = threadIdx.x;
  int wave = tid >> 5, lane = tid & 31, n = lane & 15, hi = lane >> 4;
  const float* xg = x + (size_t)b * 1024 * 7;
  for (int i = tid; i < 1024 * 7; i += 256) sxb[i / 7][i % 7] = f2bfu(xg[i]);
  for (int i = tid; i < 32 * 64; i += 256) {
    int k = i >> 6, o = i & 63;
    Bw1[k][o] = (k < 14) ? f2bfu(ecw1[o * 14 + k]) : (unsigned short)0;
  }
  for (int i = tid; i < 64 * 128; i += 256) {
    int k = i >> 7, o = i & 127;
    Bw2[k][o] = f2bfu(ecw2[o * 64 + k]);
  }
  if (tid < 64) { sc1[tid] = bnp1[tid]; sh1[tid] = bnp1[64 + tid]; }
  if (tid < 128) { sc2[tid] = bnp2[tid]; sh2[tid] = bnp2[128 + tid]; }
  __syncthreads();
  v16bf w1f[4];
  #pragma unroll
  for (int t = 0; t < 4; t++)
    #pragma unroll
    for (int j = 0; j < 16; j++) w1f[t][j] = usb(Bw1[hi * 16 + j][t * 16 + n]);
  v16bf w2f[2][8];
  #pragma unroll
  for (int chunk = 0; chunk < 2; chunk++)
    #pragma unroll
    for (int nt = 0; nt < 8; nt++)
      #pragma unroll
      for (int j = 0; j < 16; j++) w2f[chunk][nt][j] = usb(Bw2[chunk * 32 + hi * 16 + j][nt * 16 + n]);
  const int* idxb = gidx + (size_t)b * 1024 * 16;
  for (int p = wave * 128; p < wave * 128 + 128; p++) {
    int jn = idxb[p * 16 + n];
    v16bf a;
    #pragma unroll
    for (int j = 0; j < 16; j++) {
      int kk = hi * 16 + j;
      unsigned short v;
      if (kk < 7) v = sxb[p][kk];
      else if (kk < 14) v = f2bfu(bfu2f(sxb[jn][kk - 7]) - bfu2f(sxb[p][kk - 7]));
      else v = 0;
      a[j] = usb(v);
    }
    #pragma unroll
    for (int t = 0; t < 4; t++) {
      v8f c = vzero();
      c = wmma_bf16(a, w1f[t], c);
      int ch = t * 16 + n;
      float sc = sc1[ch], sh = sh1[ch];
      int chunk = ch >> 5, kk = ch & 31;
      #pragma unroll
      for (int r = 0; r < 8; r++) {
        int M = r + 8 * hi;
        h1f[wave][chunk][M + 16 * (kk >> 4)][kk & 15] = f2bfu(gelu_f(c[r] * sc + sh));
      }
    }
    v8f acc[8];
    #pragma unroll
    for (int t = 0; t < 8; t++) acc[t] = vzero();
    #pragma unroll
    for (int chunk = 0; chunk < 2; chunk++) {
      v16bf a2 = ld_frag_lds(&h1f[wave][chunk][lane][0]);
      #pragma unroll
      for (int nt = 0; nt < 8; nt++) acc[nt] = wmma_bf16(a2, w2f[chunk][nt], acc[nt]);
    }
    #pragma unroll
    for (int nt = 0; nt < 8; nt++) {
      int ch = nt * 16 + n;
      float sc = sc2[ch], sh = sh2[ch];
      float mx = -3.4e38f;
      #pragma unroll
      for (int r = 0; r < 8; r++) mx = fmaxf(mx, gelu_f(acc[nt][r] * sc + sh));
      float om = __shfl_xor(mx, 16);
      mx = fmaxf(mx, om);
      if (hi == 0) localf[((size_t)b * 1024 + p) * 128 + ch] = f2bfu(mx);
    }
  }
}

// ---------------------------------------------------------------------------
// K7: per-point MLP (135->128 LN gelu ->256 LN gelu) + max/mean pool over P.
// A tiles and the h1 restage live in fragment-order LDS; B fragments are
// fragment-contiguous in global ws (2x global_load_b128 each).
// ---------------------------------------------------------------------------
__global__ __launch_bounds__(256) void k_pmlp(const float* __restrict__ x, const unsigned short* __restrict__ localf,
                                              const unsigned short* __restrict__ Wp1f, const unsigned short* __restrict__ Wp2f,
                                              const float* __restrict__ pmb1, const float* __restrict__ l1g, const float* __restrict__ l1b,
                                              const float* __restrict__ pmb2, const float* __restrict__ l2g, const float* __restrict__ l2b,
                                              float* __restrict__ pooled) {
  __shared__ alignas(32) unsigned short Af[8][5][32][16];  // A frags; chunks 0..3 reused for h1
  __shared__ float sl1g[128], sl1b[128], sb1[128];
  __shared__ float sl2g[256], sl2b[256], sb2[256];
  __shared__ float pmax[8][256], psum[8][256];
  int bt = blockIdx.x, tid = threadIdx.x;
  int wave = tid >> 5, lane = tid & 31, n = lane & 15, hi = lane >> 4;
  if (tid < 128) { sl1g[tid] = l1g[tid]; sl1b[tid] = l1b[tid]; sb1[tid] = pmb1[tid]; }
  for (int i = tid; i < 256; i += 256) { sl2g[i] = l2g[i]; sl2b[i] = l2b[i]; sb2[i] = pmb2[i]; }
  __syncthreads();
  float accmax[16], accsum[16];
  #pragma unroll
  for (int t = 0; t < 16; t++) { accmax[t] = -3.4e38f; accsum[t] = 0.f; }
  for (int tile = 0; tile < 8; tile++) {
    int prow0 = wave * 128 + tile * 16;
    for (int e = lane; e < 16 * 160; e += 32) {
      int rr = e / 160, cc = e % 160;
      size_t rowg = (size_t)bt * 1024 + prow0 + rr;
      unsigned short v;
      if (cc < 128) v = localf[rowg * 128 + cc];
      else if (cc < 135) v = f2bfu(x[rowg * 7 + cc - 128]);
      else v = 0;
      Af[wave][cc >> 5][rr + 16 * ((cc & 31) >> 4)][cc & 15] = v;
    }
    v16bf af[5];
    #pragma unroll
    for (int kc = 0; kc < 5; kc++) af[kc] = ld_frag_lds(&Af[wave][kc][lane][0]);
    v8f y1[8];
    #pragma unroll
    for (int nt = 0; nt < 8; nt++) {
      float bias = sb1[nt * 16 + n];
      #pragma unroll
      for (int r = 0; r < 8; r++) y1[nt][r] = bias;
    }
    #pragma unroll
    for (int nt = 0; nt < 8; nt++)
      #pragma unroll
      for (int kc = 0; kc < 5; kc++) {
        v16bf bb = ld_frag_g(Wp1f + ((size_t)(kc * 8 + nt) * 32 + lane) * 16);
        y1[nt] = wmma_bf16(af[kc], bb, y1[nt]);
      }
    #pragma unroll
    for (int r = 0; r < 8; r++) {  // LN over 128 per row (r, hi)
      float s = 0.f, q = 0.f;
      #pragma unroll
      for (int nt = 0; nt < 8; nt++) { float v = y1[nt][r]; s += v; q += v * v; }
      #pragma unroll
      for (int off = 1; off < 16; off <<= 1) { s += __shfl_xor(s, off); q += __shfl_xor(q, off); }
      float mean = s * (1.f / 128.f);
      float var = q * (1.f / 128.f) - mean * mean;
      float rstd = rsqrtf(var + 1e-5f);
      #pragma unroll
      for (int nt = 0; nt < 8; nt++) {
        int ch = nt * 16 + n;
        y1[nt][r] = gelu_f((y1[nt][r] - mean) * rstd * sl1g[ch] + sl1b[ch]);
      }
    }
    #pragma unroll
    for (int nt = 0; nt < 8; nt++) {
      int ch = nt * 16 + n;
      int chunk = ch >> 5, kk = ch & 31;
      #pragma unroll
      for (int r = 0; r < 8; r++)
        Af[wave][chunk][(r + 8 * hi) + 16 * (kk >> 4)][kk & 15] = f2bfu(y1[nt][r]);
    }
    v16bf a2[4];
    #pragma unroll
    for (int kc = 0; kc < 4; kc++) a2[kc] = ld_frag_lds(&Af[wave][kc][lane][0]);
    v8f y2[16];
    #pragma unroll
    for (int nt = 0; nt < 16; nt++) {
      float bias = sb2[nt * 16 + n];
      #pragma unroll
      for (int r = 0; r < 8; r++) y2[nt][r] = bias;
    }
    #pragma unroll
    for (int nt = 0; nt < 16; nt++)
      #pragma unroll
      for (int kc = 0; kc < 4; kc++) {
        v16bf bb = ld_frag_g(Wp2f + ((size_t)(kc * 16 + nt) * 32 + lane) * 16);
        y2[nt] = wmma_bf16(a2[kc], bb, y2[nt]);
      }
    #pragma unroll
    for (int r = 0; r < 8; r++) {  // LN over 256
      float s = 0.f, q = 0.f;
      #pragma unroll
      for (int nt = 0; nt < 16; nt++) { float v = y2[nt][r]; s += v; q += v * v; }
      #pragma unroll
      for (int off = 1; off < 16; off <<= 1) { s += __shfl_xor(s, off); q += __shfl_xor(q, off); }
      float mean = s * (1.f / 256.f);
      float var = q * (1.f / 256.f) - mean * mean;
      float rstd = rsqrtf(var + 1e-5f);
      #pragma unroll
      for (int nt = 0; nt < 16; nt++) {
        int ch = nt * 16 + n;
        y2[nt][r] = gelu_f((y2[nt][r] - mean) * rstd * sl2g[ch] + sl2b[ch]);
      }
    }
    #pragma unroll
    for (int nt = 0; nt < 16; nt++) {
      float m = -3.4e38f, sv = 0.f;
      #pragma unroll
      for (int r = 0; r < 8; r++) { float v = y2[nt][r]; m = fmaxf(m, v); sv += v; }
      accmax[nt] = fmaxf(accmax[nt], m);
      accsum[nt] += sv;
    }
  }
  #pragma unroll
  for (int nt = 0; nt < 16; nt++) {
    float om = __shfl_xor(accmax[nt], 16); accmax[nt] = fmaxf(accmax[nt], om);
    float os = __shfl_xor(accsum[nt], 16); accsum[nt] += os;
  }
  if (hi == 0)
    #pragma unroll
    for (int nt = 0; nt < 16; nt++) { pmax[wave][nt * 16 + n] = accmax[nt]; psum[wave][nt * 16 + n] = accsum[nt]; }
  __syncthreads();
  if (tid < 256) {
    int ch = tid;
    float m = pmax[0][ch], s = psum[0][ch];
    #pragma unroll
    for (int w = 1; w < 8; w++) { m = fmaxf(m, pmax[w][ch]); s += psum[w][ch]; }
    pooled[(size_t)bt * 512 + ch] = m;
    pooled[(size_t)bt * 512 + 256 + ch] = s * (1.f / 1024.f);
  }
}

// ---------------------------------------------------------------------------
// Tail: proj, 4x (conv1d T=32 + BN over (b,t) + gelu [+res]), maxT, MLP head
// ---------------------------------------------------------------------------
__global__ __launch_bounds__(256) void k_proj(const float* __restrict__ pooled, const float* __restrict__ pw,
                                              const float* __restrict__ pb, float* __restrict__ h0) {
  __shared__ float row[512];
  int pos = blockIdx.x, o = threadIdx.x;
  for (int i = o; i < 512; i += 256) row[i] = pooled[(size_t)pos * 512 + i];
  __syncthreads();
  float acc = pb[o];
  for (int c = 0; c < 512; c++) acc += row[c] * pw[(size_t)o * 512 + c];
  h0[(size_t)pos * 256 + o] = acc;
}

__global__ __launch_bounds__(256) void k_conv(const float* __restrict__ hin, const float* __restrict__ w,
                                              const float* __restrict__ cb, const float* __restrict__ bg,
                                              const float* __restrict__ bb, float* __restrict__ hout, int residual) {
  int wave = threadIdx.x >> 5, lane = threadIdx.x & 31;
  int o = blockIdx.x * 8 + wave;  // one wave per output channel, 64 positions
  float y[2];
  #pragma unroll
  for (int s = 0; s < 2; s++) {
    int pos = lane * 2 + s;
    int b = pos >> 5, t = pos & 31;
    float acc = cb[o];
    #pragma unroll
    for (int dt = 0; dt < 3; dt++) {
      int tt = t + dt - 1;
      if (tt >= 0 && tt < 32) {
        const float* hr = hin + ((size_t)(b * 32 + tt)) * 256;
        const float* wr = w + (size_t)o * 256 * 3 + dt;
        for (int c = 0; c < 256; c++) acc += hr[c] * wr[c * 3];
      }
    }
    y[s] = acc;
  }
  float s = y[0] + y[1], q = y[0] * y[0] + y[1] * y[1];
  #pragma unroll
  for (int off = 1; off < 32; off <<= 1) { s += __shfl_xor(s, off); q += __shfl_xor(q, off); }
  float mean = s * (1.f / 64.f);
  float var = q * (1.f / 64.f) - mean * mean;
  float sc = bg[o] * rsqrtf(var + 1e-5f), sh = bb[o] - mean * sc;
  #pragma unroll
  for (int ss = 0; ss < 2; ss++) {
    int pos = lane * 2 + ss;
    float v = gelu_f(y[ss] * sc + sh);
    if (residual) v += hin[(size_t)pos * 256 + o];
    hout[(size_t)pos * 256 + o] = v;
  }
}

__global__ __launch_bounds__(256) void k_head1(const float* __restrict__ hfin, const float* __restrict__ h1w,
                                               const float* __restrict__ h1b, float* __restrict__ hid) {
  __shared__ float hm[512];
  int tid = threadIdx.x;
  for (int i = tid; i < 512; i += 256) {
    int b = i >> 8, o = i & 255;
    float m = -3.4e38f;
    for (int t = 0; t < 32; t++) m = fmaxf(m, hfin[(size_t)(b * 32 + t) * 256 + o]);
    hm[i] = m;
  }
  __syncthreads();
  int b = tid >> 7, j = tid & 127;
  float acc = h1b[j];
  for (int c = 0; c < 256; c++) acc += hm[b * 256 + c] * h1w[(size_t)j * 256 + c];
  hid[tid] = gelu_f(acc);
}

__global__ __launch_bounds__(64) void k_head2(const float* __restrict__ hid, const float* __restrict__ h2w,
                                              const float* __restrict__ h2b, float* __restrict__ out) {
  int t = threadIdx.x;
  if (t < 50) {
    int b = t / 25, nn = t % 25;
    float acc = h2b[nn];
    for (int j = 0; j < 128; j++) acc += hid[b * 128 + j] * h2w[(size_t)nn * 128 + j];
    out[t] = acc;
  }
}

// prepack pm weights as bf16 in B-fragment-contiguous order:
// Wp1f[((kc*8+nt)*32 + lane)*16 + j]  = pm_w1[N=nt*16+(lane&15)][K=kc*32+(lane>>4)*16+j]
// Wp2f[((kc*16+nt)*32 + lane)*16 + j] = pm_w2[N][K]
__global__ __launch_bounds__(256) void k_prep(const float* __restrict__ pmw1, const float* __restrict__ pmw2,
                                              unsigned short* __restrict__ Wp1f, unsigned short* __restrict__ Wp2f) {
  int t = blockIdx.x * 256 + threadIdx.x;
  if (t < 40 * 512) {
    int tileI = t >> 9, rem = t & 511;
    int lane = rem >> 4, j = rem & 15;
    int kc = tileI / 8, nt = tileI % 8;
    int k = kc * 32 + (lane >> 4) * 16 + j;
    int o = nt * 16 + (lane & 15);
    Wp1f[t] = (k < 135) ? f2bfu(pmw1[(size_t)o * 135 + k]) : (unsigned short)0;
  } else {
    int u = t - 40 * 512;
    if (u < 64 * 512) {
      int tileI = u >> 9, rem = u & 511;
      int lane = rem >> 4, j = rem & 15;
      int kc = tileI >> 4, nt = tileI & 15;
      int k = kc * 32 + (lane >> 4) * 16 + j;
      int o = nt * 16 + (lane & 15);
      Wp2f[u] = f2bfu(pmw2[(size_t)o * 128 + k]);
    }
  }
}

extern "C" void kernel_launch(void* const* d_in, const int* in_sizes, int n_in,
                              void* d_out, int out_size, void* d_ws, size_t ws_size,
                              hipStream_t stream) {
  const float* x     = (const float*)d_in[0];
  const float* ec_w1 = (const float*)d_in[1];
  const float* ec_g1 = (const float*)d_in[2];
  const float* ec_b1 = (const float*)d_in[3];
  const float* ec_w2 = (const float*)d_in[4];
  const float* ec_g2 = (const float*)d_in[5];
  const float* ec_b2 = (const float*)d_in[6];
  const float* pm_w1 = (const float*)d_in[7];
  const float* pm_b1 = (const float*)d_in[8];
  const float* ln1_g = (const float*)d_in[9];
  const float* ln1_b = (const float*)d_in[10];
  const float* pm_w2 = (const float*)d_in[11];
  const float* pm_b2 = (const float*)d_in[12];
  const float* ln2_g = (const float*)d_in[13];
  const float* ln2_b = (const float*)d_in[14];
  const float* prj_w = (const float*)d_in[15];
  const float* prj_b = (const float*)d_in[16];
  const float* c1w = (const float*)d_in[17]; const float* c1b = (const float*)d_in[18];
  const float* g1  = (const float*)d_in[19]; const float* b1  = (const float*)d_in[20];
  const float* c2w = (const float*)d_in[21]; const float* c2b = (const float*)d_in[22];
  const float* g2  = (const float*)d_in[23]; const float* b2  = (const float*)d_in[24];
  const float* c3w = (const float*)d_in[25]; const float* c3b = (const float*)d_in[26];
  const float* g3  = (const float*)d_in[27]; const float* b3  = (const float*)d_in[28];
  const float* c4w = (const float*)d_in[29]; const float* c4b = (const float*)d_in[30];
  const float* g4  = (const float*)d_in[31]; const float* b4  = (const float*)d_in[32];
  const float* h1w = (const float*)d_in[33]; const float* h1b = (const float*)d_in[34];
  const float* h2w = (const float*)d_in[35]; const float* h2b = (const float*)d_in[36];

  char* wsb = (char*)d_ws;
  size_t off = 0;
  auto take = [&](size_t bytes) { size_t o = off; off = (off + bytes + 255) & ~(size_t)255; return o; };
  int*            g_idx  = (int*)(wsb + take((size_t)64 * 1024 * 16 * 4));
  float*          part1  = (float*)(wsb + take((size_t)64 * 210 * 4));
  float*          bnp1   = (float*)(wsb + take(128 * 4));
  float*          part2  = (float*)(wsb + take((size_t)512 * 4160 * 4));
  float*          bnp2   = (float*)(wsb + take(256 * 4));
  unsigned short* localf = (unsigned short*)(wsb + take((size_t)64 * 1024 * 128 * 2));
  unsigned short* Wp1f   = (unsigned short*)(wsb + take((size_t)40 * 512 * 2));
  unsigned short* Wp2f   = (unsigned short*)(wsb + take((size_t)64 * 512 * 2));
  float*          pooled = (float*)(wsb + take((size_t)64 * 512 * 4));
  float*          hb0    = (float*)(wsb + take((size_t)64 * 256 * 4));
  float*          hb1    = (float*)(wsb + take((size_t)64 * 256 * 4));
  float*          hid    = (float*)(wsb + take(256 * 4));
  (void)ws_size; (void)n_in; (void)in_sizes; (void)out_size;

  k_prep  <<<208, 256, 0, stream>>>(pm_w1, pm_w2, Wp1f, Wp2f);
  k_knn   <<<64, 256, 0, stream>>>(x, g_idx);
  k_stats1<<<64, 256, 0, stream>>>(x, g_idx, part1);
  k_fin1  <<<1, 256, 0, stream>>>(part1, ec_w1, ec_g1, ec_b1, bnp1);
  k_stats2<<<64, 256, 0, stream>>>(x, g_idx, ec_w1, bnp1, part2);
  k_fin2  <<<1, 256, 0, stream>>>(part2, ec_w2, ec_g2, ec_b2, bnp2);
  k_edge  <<<64, 256, 0, stream>>>(x, g_idx, ec_w1, ec_w2, bnp1, bnp2, localf);
  k_pmlp  <<<64, 256, 0, stream>>>(x, localf, Wp1f, Wp2f, pm_b1, ln1_g, ln1_b, pm_b2, ln2_g, ln2_b, pooled);
  k_proj  <<<64, 256, 0, stream>>>(pooled, prj_w, prj_b, hb0);
  k_conv  <<<32, 256, 0, stream>>>(hb0, c1w, c1b, g1, b1, hb1, 0);
  k_conv  <<<32, 256, 0, stream>>>(hb1, c2w, c2b, g2, b2, hb0, 1);
  k_conv  <<<32, 256, 0, stream>>>(hb0, c3w, c3b, g3, b3, hb1, 1);
  k_conv  <<<32, 256, 0, stream>>>(hb1, c4w, c4b, g4, b4, hb0, 1);
  k_head1 <<<1, 256, 0, stream>>>(hb0, h1w, h1b, hid);
  k_head2 <<<1, 64, 0, stream>>>(hid, h2w, h2b, (float*)d_out);
}
